// MultiLayerShiftVaryingSystem_57595511439511
// MI455X (gfx1250) — matrix operand order repaired and checked by reading, not compiled, not run on hardware
//
#include <hip/hip_runtime.h>
#include <hip/hip_bf16.h>

// Problem constants (from the reference)
#define H      256
#define W      256
#define BC     24          // b*c = 8*3
#define NPSF   8
#define BW     15          // band width
#define BHALF  7
#define KS     7
#define KPAD   3           // KS/2
#define DLOW   128         // low-res weight grid (H/2)
#define NFILT  (NPSF * KS * KS)   // 392 filter taps

// Tiling
#define TH     8           // output tile height
#define TW     8           // output tile width
#define RH     (TH + 2*KPAD)   // 14: mid region height (conv halo)
#define RW     (TW + 2*KPAD)   // 14
#define NPIX   (RH * RW)       // 196 mid pixels per tile
#define BLOCK  128             // 4 wave32 waves
#define NWAVE  (BLOCK / 32)

typedef __attribute__((ext_vector_type(16))) _Float16 v16h;
typedef __attribute__((ext_vector_type(8)))  float    v8f;
typedef int v4i __attribute__((vector_size(16)));   // matches async-LDS builtin param

#define GLOBAL_AS __attribute__((address_space(1)))
#define LDS_AS    __attribute__((address_space(3)))

#if defined(__has_builtin)
#  if __has_builtin(__builtin_amdgcn_global_load_async_to_lds_b128)
#    define HAVE_ASYNC_LDS 1
#  endif
#endif

__global__ __launch_bounds__(BLOCK)
void msvs_fused_wmma_kernel(const float* __restrict__ x,
                            const float* __restrict__ pw,
                            const float* __restrict__ filt,
                            float* __restrict__ out)
{
    // mid tile: [region pixel][bc within group (M, 16)][psf (N, 8)] as f16
    __shared__ _Float16 smid[NPIX][16][NPSF];     // 50176 B
    __shared__ float    sfraw[NFILT];             // raw filter staging
    __shared__ float    sfltT[KS][KS][NPSF];      // transposed: taps contiguous in n

    const int tid    = threadIdx.x;
    const int lane   = tid & 31;
    const int wave   = tid >> 5;
    const int x0     = blockIdx.x * TW;
    const int y0     = blockIdx.y * TH;
    const int bcgrp  = blockIdx.z;          // 0: bc 0..15, 1: bc 16..23 (+pad)

    // ---- stage filters into LDS via the CDNA5 async global->LDS path ----
#ifdef HAVE_ASYNC_LDS
    if (tid < NFILT / 4) {                  // 98 lanes x b128 = 392 floats
        GLOBAL_AS v4i* g = (GLOBAL_AS v4i*)filt + tid;
        LDS_AS    v4i* l = (LDS_AS v4i*)sfraw + tid;
        __builtin_amdgcn_global_load_async_to_lds_b128(g, l, 0, 0);
    }
    asm volatile("s_wait_asynccnt 0x0" ::: "memory");
#else
    for (int i = tid; i < NFILT; i += BLOCK)
        sfraw[i] = filt[i];
#endif
    __syncthreads();
    // transpose [n][ky][kx] -> [ky][kx][n] so PSF reduction reads contiguously
    for (int i = tid; i < NFILT; i += BLOCK) {
        const int n = i / (KS * KS);
        const int r = i - n * (KS * KS);
        sfltT[r / KS][r % KS][n] = sfraw[i];
    }

    // Warm L2 for this tile's x rows (global_prefetch path).
    if (lane == 0) {
        int gy = y0 + (int)wave * 2;
        if (gy < H) {
            const float* p = x + ((size_t)(bcgrp * 16) * H + gy) * W + x0;
            __builtin_prefetch(p, 0, 1);
        }
    }

    // ---------------- Phase 1: per-pixel batched GEMM via WMMA ----------------
    // D[bc(16), n(16)] = A[bc(16), d(32)] x B[d(32), n(16)], K padded 15->32,
    // N padded 8->16. One v_wmma_f32_16x16x32_f16 per mid pixel per wave.
    const int mrow = lane & 15;                 // M row this lane carries
    const int bcg  = bcgrp * 16 + mrow;         // global bc image
    const bool rowv = (bcg < BC);
    const int kbA  = (lane < 16) ? 0 : 8;       // half-wave select; A K base is 2*kbA = 16h
    const int ncol = lane & 15;                 // N column for B and D

    for (int p = wave; p < NPIX; p += NWAVE) {
        const int ry = p / RW;
        const int rx = p - ry * RW;
        const int gy = y0 - KPAD + ry;          // mid pixel coords in image
        const int gx = x0 - KPAD + rx;

        if (gy < 0 || gy >= H || gx < 0 || gx >= W) {
            // conv zero-padding region of mid
            for (int i = lane; i < 16 * NPSF; i += 32)
                ((_Float16*)smid[p])[i] = (_Float16)0.0f;
            continue;
        }

        // --- A fragment: x band values. K = d = 16h + e; lanes 16..31 map to K>=16 -> 0.
        v16h a;
        const float* xrow = x + ((size_t)(rowv ? bcg : 0) * H + gy) * W;
        #pragma unroll
        for (int e = 0; e < 16; ++e) {
            const int d   = 2 * kbA + e;        // 0..15 or 16..31
            const int col = gx + d - BHALF;     // width-only zero pad
            float v = 0.0f;
            if (rowv && d < BW && col >= 0 && col < W) v = xrow[col];
            a[e]     = (_Float16)v;
        }

        // --- B fragment: bilinear-upsampled band weights, lane n in 0..7.
        // Half-pixel centers: src = 0.5*dst - 0.25, edges clamped.
        float wv[16];
        #pragma unroll
        for (int e = 0; e < 16; ++e) wv[e] = 0.0f;
        if (lane < NPSF) {
            const float sy = 0.5f * (float)gy - 0.25f;
            const float sx = 0.5f * (float)gx - 0.25f;
            int iy0 = (int)floorf(sy);
            int ix0 = (int)floorf(sx);
            const float fy = sy - (float)iy0;
            const float fx = sx - (float)ix0;
            int iy1 = iy0 + 1, ix1 = ix0 + 1;
            iy0 = min(max(iy0, 0), DLOW - 1);
            iy1 = min(max(iy1, 0), DLOW - 1);
            ix0 = min(max(ix0, 0), DLOW - 1);
            ix1 = min(max(ix1, 0), DLOW - 1);
            const float w00 = (1.0f - fy) * (1.0f - fx);
            const float w01 = (1.0f - fy) * fx;
            const float w10 = fy * (1.0f - fx);
            const float w11 = fy * fx;
            const float* base = pw + (size_t)lane * (DLOW * DLOW * BW);
            const float* r00 = base + ((size_t)iy0 * DLOW + ix0) * BW;
            const float* r01 = base + ((size_t)iy0 * DLOW + ix1) * BW;
            const float* r10 = base + ((size_t)iy1 * DLOW + ix0) * BW;
            const float* r11 = base + ((size_t)iy1 * DLOW + ix1) * BW;
            #pragma unroll
            for (int d = 0; d < BW; ++d)
                wv[d] = w00 * r00[d] + w01 * r01[d] + w10 * r10[d] + w11 * r11[d];
        }
        // Lanes 0..15 carry K=0..15 (only lanes 0..7 nonzero => N pad);
        // lanes 16..31 carry K=16..31 which is entirely K-pad (zero).
        v16h b;
        #pragma unroll
        for (int e = 0; e < 16; ++e)
            b[e] = (lane < 16) ? (_Float16)wv[e] : (_Float16)0.0f;

        v8f c = {};
        c = __builtin_amdgcn_wmma_f32_16x16x32_f16(
                /*neg_a=*/false, a, /*neg_b=*/false, b,
                /*c_mod=*/(short)0, c, /*reuse_a=*/false, /*reuse_b=*/false);

        // D layout: lane<16 -> N=lane, M=r; lane>=16 -> N=lane-16, M=8+r.
        if (ncol < NPSF) {
            const int mbase = (lane < 16) ? 0 : 8;
            #pragma unroll
            for (int r = 0; r < 8; ++r)
                smid[p][mbase + r][ncol] = (_Float16)c[r];
        }
    }

    __syncthreads();

    // ------- Phase 2: fused depthwise 7x7 conv + PSF sum, row-blocked -------
    // thread = (bc, oy); 8 ox accumulators; each halo pixel's 8-PSF vector is
    // loaded once per (ky, rxx) as a b128 LDS load (4x fewer ds ops than
    // per-output gathering).
    {
        const int bcl = tid >> 3;               // 0..15
        const int oy  = tid & 7;
        const int bcO = bcgrp * 16 + bcl;
        if (bcO < BC) {
            float acc[TW];
            #pragma unroll
            for (int i = 0; i < TW; ++i) acc[i] = 0.0f;

            for (int ky = 0; ky < KS; ++ky) {
                const int ry = oy + ky;
                for (int rxx = 0; rxx < RW; ++rxx) {
                    const _Float16* m = smid[ry * RW + rxx][bcl];
                    float vn[NPSF];
                    #pragma unroll
                    for (int n = 0; n < NPSF; ++n) vn[n] = (float)m[n];
                    #pragma unroll
                    for (int ox = 0; ox < TW; ++ox) {
                        const int kx = rxx - ox;
                        if (kx >= 0 && kx < KS) {
                            const float* fv = &sfltT[ky][kx][0];
                            float s = 0.0f;
                            #pragma unroll
                            for (int n = 0; n < NPSF; ++n) s += vn[n] * fv[n];
                            acc[ox] += s;
                        }
                    }
                }
            }
            float* orow = out + ((size_t)bcO * H + (y0 + oy)) * W + x0;
            #pragma unroll
            for (int ox = 0; ox < TW; ++ox) orow[ox] = acc[ox];
        }
    }
}

extern "C" void kernel_launch(void* const* d_in, const int* in_sizes, int n_in,
                              void* d_out, int out_size, void* d_ws, size_t ws_size,
                              hipStream_t stream)
{
    (void)in_sizes; (void)n_in; (void)d_ws; (void)ws_size; (void)out_size;
    const float* x    = (const float*)d_in[0];   // (8,3,256,256)
    const float* pw   = (const float*)d_in[1];   // (8, 128*128, 15)
    const float* filt = (const float*)d_in[2];   // (8,1,7,7)
    float* out = (float*)d_out;                  // (8,3,256,256)

    dim3 grid(W / TW, H / TH, 2);                // 32 x 32 tiles x 2 bc-groups
    dim3 block(BLOCK);
    msvs_fused_wmma_kernel<<<grid, block, 0, stream>>>(x, pw, filt, out);
}
